// LRM_4209067950423
// MI455X (gfx1250) — compile-verified, matrix operand order checked by reading, not run on hardware
//
#include <hip/hip_runtime.h>
#include <math.h>

#define D 256
#define D4 (D / 4)

typedef float  v2f  __attribute__((ext_vector_type(2)));
typedef float  v8f  __attribute__((ext_vector_type(8)));
typedef __bf16 v16bf __attribute__((ext_vector_type(16)));

// ---------------------------------------------------------------------------
// float atomic-max via integer punning (works for mixed signs, init = -inf)
// ---------------------------------------------------------------------------
__device__ __forceinline__ void atomicMaxF(float* addr, float val) {
    if (!(val < 0.0f)) atomicMax((int*)addr, __float_as_int(val));
    else               atomicMin((unsigned int*)addr, __float_as_uint(val));
}

// ---------------------------------------------------------------------------
// init: segment-max = -inf, segment-sum = 0
// ---------------------------------------------------------------------------
__global__ void init_nodes(float* mp, float* md, float* ms,
                           float* sp, float* sd, float* ss, int N) {
    int i = blockIdx.x * blockDim.x + threadIdx.x;
    if (i < N) {
        mp[i] = -INFINITY; md[i] = -INFINITY; ms[i] = -INFINITY;
        sp[i] = 0.0f;      sd[i] = 0.0f;      ss[i] = 0.0f;
    }
}

// ---------------------------------------------------------------------------
// one wave (32 lanes) per edge: 3 channel dot products over D=256,
// wave reduction, store score + atomic segment-max per destination
// ---------------------------------------------------------------------------
__global__ void edge_scores(const float* __restrict__ pos, const float* __restrict__ dis,
                            const float* __restrict__ syn,
                            const int* __restrict__ src, const int* __restrict__ dst, int E,
                            float* __restrict__ s_pos, float* __restrict__ s_dis,
                            float* __restrict__ s_syn,
                            float* __restrict__ m_pos, float* __restrict__ m_dis,
                            float* __restrict__ m_syn) {
    int e    = (blockIdx.x * blockDim.x + threadIdx.x) >> 5;
    int lane = threadIdx.x & 31;
    if (e >= E) return;
    int u = src[e], v = dst[e];

    const float4* pu = (const float4*)pos + (size_t)u * D4 + lane * 2;
    const float4* pv = (const float4*)pos + (size_t)v * D4 + lane * 2;
    const float4* du = (const float4*)dis + (size_t)u * D4 + lane * 2;
    const float4* dv = (const float4*)dis + (size_t)v * D4 + lane * 2;
    const float4* su = (const float4*)syn + (size_t)u * D4 + lane * 2;
    const float4* sv = (const float4*)syn + (size_t)v * D4 + lane * 2;

    float dp, dd, dsn;
    {
        float4 a0 = pu[0], a1 = pu[1], b0 = pv[0], b1 = pv[1];
        dp = a0.x*b0.x + a0.y*b0.y + a0.z*b0.z + a0.w*b0.w
           + a1.x*b1.x + a1.y*b1.y + a1.z*b1.z + a1.w*b1.w;
    }
    {
        float4 a0 = du[0], a1 = du[1], b0 = dv[0], b1 = dv[1];
        dd = a0.x*b0.x + a0.y*b0.y + a0.z*b0.z + a0.w*b0.w
           + a1.x*b1.x + a1.y*b1.y + a1.z*b1.z + a1.w*b1.w;
    }
    {
        float4 a0 = su[0], a1 = su[1], b0 = sv[0], b1 = sv[1];
        dsn = a0.x*b0.x + a0.y*b0.y + a0.z*b0.z + a0.w*b0.w
            + a1.x*b1.x + a1.y*b1.y + a1.z*b1.z + a1.w*b1.w;
    }
#pragma unroll
    for (int off = 16; off; off >>= 1) {
        dp  += __shfl_xor(dp,  off, 32);
        dd  += __shfl_xor(dd,  off, 32);
        dsn += __shfl_xor(dsn, off, 32);
    }
    if (lane == 0) {
        s_pos[e] = dp; s_dis[e] = dd; s_syn[e] = dsn;
        atomicMaxF(&m_pos[v], dp);
        atomicMaxF(&m_dis[v], dd);
        atomicMaxF(&m_syn[v], dsn);
    }
}

// ---------------------------------------------------------------------------
// one thread per edge: stabilized exp, atomic segment-sum per destination
// ---------------------------------------------------------------------------
__global__ void edge_sums(const int* __restrict__ dst, int E,
                          const float* __restrict__ s_pos, const float* __restrict__ s_dis,
                          const float* __restrict__ s_syn,
                          const float* __restrict__ m_pos, const float* __restrict__ m_dis,
                          const float* __restrict__ m_syn,
                          float* __restrict__ sum_pos, float* __restrict__ sum_dis,
                          float* __restrict__ sum_syn) {
    int e = blockIdx.x * blockDim.x + threadIdx.x;
    if (e >= E) return;
    int v = dst[e];
    atomicAdd(&sum_pos[v], expf(s_pos[e] - m_pos[v]));
    atomicAdd(&sum_dis[v], expf(s_dis[e] - m_dis[v]));
    atomicAdd(&sum_syn[v], expf(s_syn[e] - m_syn[v]));
}

// ---------------------------------------------------------------------------
// one wave per edge: fused weight (intra-softmax + cross-softmax fractions),
// scatter-add w * h[src] into out[3][N][D] with f32 atomics (L2-resident)
// ---------------------------------------------------------------------------
__global__ void aggregate(const float* __restrict__ h,
                          const int* __restrict__ src, const int* __restrict__ dst,
                          int E, int N,
                          const float* __restrict__ s_pos, const float* __restrict__ s_dis,
                          const float* __restrict__ s_syn,
                          const float* __restrict__ m_pos, const float* __restrict__ m_dis,
                          const float* __restrict__ m_syn,
                          const float* __restrict__ sum_pos, const float* __restrict__ sum_dis,
                          const float* __restrict__ sum_syn,
                          float* __restrict__ out) {
    int e    = (blockIdx.x * blockDim.x + threadIdx.x) >> 5;
    int lane = threadIdx.x & 31;
    if (e >= E) return;
    int u = src[e], v = dst[e];

    float sp = s_pos[e], sd = s_dis[e], ss = s_syn[e];
    float wp  = expf(sp - m_pos[v]) / sum_pos[v];
    float wd  = expf(sd - m_dis[v]) / sum_dis[v];
    float wsn = expf(ss - m_syn[v]) / sum_syn[v];
    // cross channel: unstabilized per-edge softmax across the 3 scores
    float ep = expf(sp), ed = expf(sd), es = expf(ss);
    float inv = 1.0f / (ep + ed + es);
    wp += ep * inv; wd += ed * inv; wsn += es * inv;

    const float4* hu = (const float4*)h + (size_t)u * D4 + lane * 2;
    float4 h0 = hu[0], h1 = hu[1];
    float hv[8] = {h0.x, h0.y, h0.z, h0.w, h1.x, h1.y, h1.z, h1.w};

    size_t nd  = (size_t)N * D;
    float* o_p = out + (size_t)v * D + lane * 8;
    float* o_d = o_p + nd;
    float* o_s = o_d + nd;
#pragma unroll
    for (int t = 0; t < 8; ++t) {
        atomicAdd(o_p + t, wp  * hv[t]);
        atomicAdd(o_d + t, wd  * hv[t]);
        atomicAdd(o_s + t, wsn * hv[t]);
    }
}

// ---------------------------------------------------------------------------
// WMMA bias GEMM: out[c][i][j] += (X_c @ W_c^T)[i][j] + b_c[j] + h[i][j]
// one wave per 16x64 output strip; 4 independent accumulator chains.
// Primary path: V_WMMA_F32_16X16X4_F32 (exact f32).
// Fallback:     V_WMMA_F32_16X16X32_BF16 with hi/lo split (~f32 accuracy).
// ---------------------------------------------------------------------------
__global__ void bias_gemm(const float* pos, const float* dis, const float* syn,
                          const float* Wp, const float* Wd, const float* Wsn,
                          const float* bp, const float* bd, const float* bs,
                          const float* h, float* out, int N)
#if !defined(__AMDGCN__)
{ /* host pass: stub body only */ }
#elif __has_builtin(__builtin_amdgcn_wmma_f32_16x16x4_f32)
{
    const int lane = threadIdx.x & 31;
    const int half = lane >> 4;     // which 16-lane half
    const int ln   = lane & 15;     // row (A) / column (B,C,D) index
    const int c    = blockIdx.z;

    const float* X    = (c == 0) ? pos : (c == 1) ? dis : syn;
    const float* W    = (c == 0) ? Wp  : (c == 1) ? Wd  : Wsn;
    const float* bias = (c == 0) ? bp  : (c == 1) ? bd  : bs;

    const int i0 = blockIdx.x * 16;
    const int j0 = blockIdx.y * 64;

    int ai = i0 + ln; if (ai >= N) ai = N - 1;          // clamp load row
    const float* arow = X + (size_t)ai * D;             // A row m=ln
    const float* br0  = W + (size_t)(j0 + ln) * D;      // B col n=ln of tile t
    const float* br1  = br0 + 16 * D;
    const float* br2  = br1 + 16 * D;
    const float* br3  = br2 + 16 * D;

    v8f acc0 = {}, acc1 = {}, acc2 = {}, acc3 = {};
#pragma unroll 8
    for (int k0 = 0; k0 < D; k0 += 4) {
        const int k = k0 + half * 2;                    // A/B f32 16x16x4 layout
        v2f a;  a[0]  = arow[k]; a[1]  = arow[k + 1];
        v2f b0; b0[0] = br0[k];  b0[1] = br0[k + 1];
        v2f b1; b1[0] = br1[k];  b1[1] = br1[k + 1];
        v2f b2; b2[0] = br2[k];  b2[1] = br2[k + 1];
        v2f b3; b3[0] = br3[k];  b3[1] = br3[k + 1];
        acc0 = __builtin_amdgcn_wmma_f32_16x16x4_f32(false, a, false, b0, (short)0, acc0, false, false);
        acc1 = __builtin_amdgcn_wmma_f32_16x16x4_f32(false, a, false, b1, (short)0, acc1, false, false);
        acc2 = __builtin_amdgcn_wmma_f32_16x16x4_f32(false, a, false, b2, (short)0, acc2, false, false);
        acc3 = __builtin_amdgcn_wmma_f32_16x16x4_f32(false, a, false, b3, (short)0, acc3, false, false);
    }

    float* outc = out + (size_t)c * N * D;
    v8f accs[4] = {acc0, acc1, acc2, acc3};
#pragma unroll
    for (int t = 0; t < 4; ++t) {
        const int jn = j0 + 16 * t + ln;
        const float bj = bias[jn];
#pragma unroll
        for (int r = 0; r < 8; ++r) {
            const int i = i0 + r + half * 8;            // C/D layout: M = r + half*8
            if (i < N) {
                const size_t idx = (size_t)i * D + jn;
                outc[idx] += accs[t][r] + bj + h[idx];
            }
        }
    }
}
#else
{
    // bf16 hi/lo split: X*W ~= Xhi*Whi + Xhi*Wlo + Xlo*Whi  (3 WMMAs / K-step)
    const int lane = threadIdx.x & 31;
    const int half = lane >> 4;
    const int ln   = lane & 15;
    const int c    = blockIdx.z;

    const float* X    = (c == 0) ? pos : (c == 1) ? dis : syn;
    const float* W    = (c == 0) ? Wp  : (c == 1) ? Wd  : Wsn;
    const float* bias = (c == 0) ? bp  : (c == 1) ? bd  : bs;

    const int i0 = blockIdx.x * 16;
    const int j0 = blockIdx.y * 64;

    int ai = i0 + ln; if (ai >= N) ai = N - 1;
    const float* arow = X + (size_t)ai * D;
    const float* brow[4];
    brow[0] = W + (size_t)(j0 + ln) * D;
    brow[1] = brow[0] + 16 * D;
    brow[2] = brow[1] + 16 * D;
    brow[3] = brow[2] + 16 * D;

    v8f acc[4] = {};
    for (int k0 = 0; k0 < D; k0 += 32) {
        // 16-bit A 16x32 layout: lane half 0 -> K = k0+{0..7, 16..23};
        //                        lane half 1 -> K = k0+{8..15, 24..31}
        const int klo = k0 + half * 8;
        const int khi = k0 + 16 + half * 8;
        v16bf ahi, alo;
#pragma unroll
        for (int t = 0; t < 8; ++t) {
            float x = arow[klo + t]; __bf16 hb = (__bf16)x;
            ahi[t] = hb; alo[t] = (__bf16)(x - (float)hb);
            float y = arow[khi + t]; __bf16 hb2 = (__bf16)y;
            ahi[8 + t] = hb2; alo[8 + t] = (__bf16)(y - (float)hb2);
        }
#pragma unroll
        for (int tt = 0; tt < 4; ++tt) {
            v16bf bhi, blo;
#pragma unroll
            for (int t = 0; t < 8; ++t) {
                float x = brow[tt][klo + t]; __bf16 hb = (__bf16)x;
                bhi[t] = hb; blo[t] = (__bf16)(x - (float)hb);
                float y = brow[tt][khi + t]; __bf16 hb2 = (__bf16)y;
                bhi[8 + t] = hb2; blo[8 + t] = (__bf16)(y - (float)hb2);
            }
            acc[tt] = __builtin_amdgcn_wmma_f32_16x16x32_bf16(false, ahi, false, bhi, (short)0, acc[tt], false, false);
            acc[tt] = __builtin_amdgcn_wmma_f32_16x16x32_bf16(false, ahi, false, blo, (short)0, acc[tt], false, false);
            acc[tt] = __builtin_amdgcn_wmma_f32_16x16x32_bf16(false, alo, false, bhi, (short)0, acc[tt], false, false);
        }
    }

    float* outc = out + (size_t)c * N * D;
#pragma unroll
    for (int t = 0; t < 4; ++t) {
        const int jn = j0 + 16 * t + ln;
        const float bj = bias[jn];
#pragma unroll
        for (int r = 0; r < 8; ++r) {
            const int i = i0 + r + half * 8;
            if (i < N) {
                const size_t idx = (size_t)i * D + jn;
                outc[idx] += acc[t][r] + bj + h[idx];
            }
        }
    }
}
#endif

// ---------------------------------------------------------------------------
extern "C" void kernel_launch(void* const* d_in, const int* in_sizes, int n_in,
                              void* d_out, int out_size, void* d_ws, size_t ws_size,
                              hipStream_t stream) {
    const float* h   = (const float*)d_in[0];
    const float* pos = (const float*)d_in[1];
    const float* dis = (const float*)d_in[2];
    const float* syn = (const float*)d_in[3];
    const float* Wp  = (const float*)d_in[4];
    const float* bp  = (const float*)d_in[5];
    const float* Wd  = (const float*)d_in[6];
    const float* bd  = (const float*)d_in[7];
    const float* Wsn = (const float*)d_in[8];
    const float* bs  = (const float*)d_in[9];
    const int*   src = (const int*)d_in[10];
    const int*   dst = (const int*)d_in[11];

    const int N = in_sizes[0] / D;
    const int E = in_sizes[10];
    float* out = (float*)d_out;

    // workspace: 3E edge scores + 6N node stats (f32)
    float* ws      = (float*)d_ws;
    float* s_pos   = ws;
    float* s_dis   = ws + (size_t)E;
    float* s_syn   = ws + 2 * (size_t)E;
    float* m_pos   = ws + 3 * (size_t)E;
    float* m_dis   = m_pos + N;
    float* m_syn   = m_dis + N;
    float* sum_pos = m_syn + N;
    float* sum_dis = sum_pos + N;
    float* sum_syn = sum_dis + N;

    hipMemsetAsync(d_out, 0, (size_t)3 * N * D * sizeof(float), stream);
    init_nodes<<<(N + 255) / 256, 256, 0, stream>>>(m_pos, m_dis, m_syn,
                                                    sum_pos, sum_dis, sum_syn, N);
    edge_scores<<<(E + 7) / 8, 256, 0, stream>>>(pos, dis, syn, src, dst, E,
                                                 s_pos, s_dis, s_syn,
                                                 m_pos, m_dis, m_syn);
    edge_sums<<<(E + 255) / 256, 256, 0, stream>>>(dst, E, s_pos, s_dis, s_syn,
                                                   m_pos, m_dis, m_syn,
                                                   sum_pos, sum_dis, sum_syn);
    aggregate<<<(E + 7) / 8, 256, 0, stream>>>(h, src, dst, E, N,
                                               s_pos, s_dis, s_syn,
                                               m_pos, m_dis, m_syn,
                                               sum_pos, sum_dis, sum_syn, out);
    dim3 g((N + 15) / 16, 4, 3);   // 16 rows x (4 tiles * 16 cols) per wave, 3 channels
    bias_gemm<<<g, 32, 0, stream>>>(pos, dis, syn, Wp, Wd, Wsn, bp, bd, bs, h, out, N);
}
